// Attention_88493506167116
// MI455X (gfx1250) — matrix operand order repaired and checked by reading, not compiled, not run on hardware
//
#include <hip/hip_runtime.h>
#include <hip/hip_bf16.h>

typedef __attribute__((ext_vector_type(16))) _Float16 v16h;
typedef __attribute__((ext_vector_type(8)))  _Float16 v8h;
typedef __attribute__((ext_vector_type(8)))  float    v8f;

#define BB 8
#define CC 256
#define NN 4096   // 64*64

union Frag16 { v16h v; v8h h[2]; };

// ---------------------------------------------------------------------------
// Tensor Data Mover support (gfx1250). Guarded so the file compiles on any
// toolchain; arity differs between ROCm 7.2 (clang-22, 5 args) and
// amdgpu-toolchain (clang-23, 6 args).
// ---------------------------------------------------------------------------
#if __has_builtin(__builtin_amdgcn_tensor_load_to_lds) && \
    __has_builtin(__builtin_amdgcn_s_wait_tensorcnt)
#define USE_TDM 1
#else
#define USE_TDM 0
#endif

#if USE_TDM
typedef unsigned int u32x4 __attribute__((ext_vector_type(4)));
typedef int          i32x4 __attribute__((ext_vector_type(4)));
typedef int          i32x8 __attribute__((ext_vector_type(8)));

// 2D tile DMA: global (row-major, stride_x elements of 2 bytes) -> LDS packed.
// D# layout per CDNA5 ISA ch.8: group0 = {count/type/lds/global addr},
// group1 = {data_size, tensor dims, tile dims, dim0 stride}.
__device__ __forceinline__ void tdm_load_2d(const void* gptr, void* lptr,
                                            unsigned tile_x, unsigned tile_y,
                                            unsigned tensor_x, unsigned tensor_y,
                                            unsigned stride_x) {
    unsigned long long ga = (unsigned long long)gptr;
    unsigned lds = (unsigned)(unsigned long long)lptr;   // low 32b of flat = LDS offset
    u32x4 g0;
    g0[0] = 1u;                                          // count=1 (valid), user mode
    g0[1] = lds;                                         // lds_addr
    g0[2] = (unsigned)(ga & 0xFFFFFFFFu);                // global_addr[31:0]
    g0[3] = (unsigned)((ga >> 32) & 0x1FFFFFFu) | (2u << 30);  // addr[56:32] | type=2
    i32x8 g1;
    g1[0] = (int)(1u << 16);                             // data_size=1 (2 bytes)
    g1[1] = (int)((tensor_x & 0xFFFFu) << 16);           // tensor_dim0[15:0]
    g1[2] = (int)((tensor_x >> 16) | ((tensor_y & 0xFFFFu) << 16)); // dim0 hi | dim1 lo
    g1[3] = (int)((tensor_y >> 16) | (tile_x << 16));    // dim1 hi | tile_dim0
    g1[4] = (int)(tile_y & 0xFFFFu);                     // tile_dim1 (tile_dim2 = 0)
    g1[5] = (int)stride_x;                               // tensor_dim0_stride[31:0]
    g1[6] = 0;
    g1[7] = 0;
    i32x4 z4 = {0, 0, 0, 0};
#if defined(__clang_major__) && (__clang_major__ >= 23)
    i32x8 z8 = {0, 0, 0, 0, 0, 0, 0, 0};
    __builtin_amdgcn_tensor_load_to_lds(g0, g1, z4, z4, z8, 0);
#else
    __builtin_amdgcn_tensor_load_to_lds(g0, g1, z4, z4, 0);
#endif
}
#endif

// ---------------------------------------------------------------------------
// f32 -> f16 elementwise convert (weights)
// ---------------------------------------------------------------------------
__global__ void f32_to_f16_kernel(const float* __restrict__ src,
                                  _Float16* __restrict__ dst, int n) {
    int i = blockIdx.x * blockDim.x + threadIdx.x;
    if (i < n) dst[i] = (_Float16)src[i];
}

// ---------------------------------------------------------------------------
// x (B,C,N) f32  ->  Xt (B,N,C) f16, via LDS 32x32 tile transpose
// ---------------------------------------------------------------------------
__global__ void transpose_convert_kernel(const float* __restrict__ x,
                                         _Float16* __restrict__ xt) {
    __shared__ float tile[32][33];
    int b  = blockIdx.z;
    int n0 = blockIdx.x * 32;
    int c0 = blockIdx.y * 32;
    const float* xb = x + (size_t)b * CC * NN;
    for (int r = threadIdx.y; r < 32; r += 8)
        tile[r][threadIdx.x] = xb[(size_t)(c0 + r) * NN + n0 + threadIdx.x];
    __syncthreads();
    _Float16* xtb = xt + (size_t)b * NN * CC;
    for (int r = threadIdx.y; r < 32; r += 8)
        xtb[(size_t)(n0 + r) * CC + c0 + threadIdx.x] = (_Float16)tile[threadIdx.x][r];
}

// ---------------------------------------------------------------------------
// Generic WMMA GEMM, K-contiguous operands. Software-pipelined via a manual
// 2x unroll with *constant-indexed* ping/pong fragment registers (runtime-
// indexed buffers defeat SROA and spill to scratch/LDS -- seen in round 2).
//   A[m,k] = A[m*lda + k]   B[k,n] = B[n*ldb + k]   D = A@B (+bias[m])
// dmode: 0 = f16 D[m*ldd+n], 1 = f16 transposed D[n*ldd+m], 2 = f32 D[m*ldd+n]
// One wave computes 16(M) x 64(N); block = 8 waves stacked in M.
// ---------------------------------------------------------------------------
__global__ void gemm_kcontig_kernel(const _Float16* __restrict__ A, size_t aBatch, int lda,
                                    const _Float16* __restrict__ Bm, size_t bBatch, int ldb,
                                    void* __restrict__ D, size_t dBatch, int ldd, int dmode,
                                    const float* __restrict__ bias,
                                    int M, int Nn, int K) {
    const int lane = threadIdx.x & 31;
    const int wid  = threadIdx.x >> 5;
    const int row  = lane & 15;
    const int hi   = lane >> 4;
    const int b    = blockIdx.z;
    const int m0   = blockIdx.y * 128 + wid * 16;
    const int n0   = blockIdx.x * 64;
    if (m0 >= M || n0 >= Nn) return;

    const _Float16* Ab = A  + aBatch * (size_t)b;
    const _Float16* Bb = Bm + bBatch * (size_t)b;
    const _Float16* a_base = Ab + (size_t)(m0 + row) * lda + hi * 8;
    const _Float16* b_base[4];
    #pragma unroll
    for (int t = 0; t < 4; ++t)
        b_base[t] = Bb + (size_t)(n0 + t * 16 + row) * ldb + hi * 8;

    v8f acc[4] = {};
    Frag16 af0, af1;
    Frag16 b0[4], b1[4];

    auto loadA = [&](Frag16& f, int k) {
        f.h[0] = *(const v8h*)(a_base + k);
        f.h[1] = *(const v8h*)(a_base + k + 16);
    };
    auto loadB = [&](Frag16 (&f)[4], int k) {
        #pragma unroll
        for (int t = 0; t < 4; ++t) {
            f[t].h[0] = *(const v8h*)(b_base[t] + k);
            f[t].h[1] = *(const v8h*)(b_base[t] + k + 16);
        }
    };

    loadA(af0, 0);
    loadB(b0, 0);
    for (int k0 = 0; k0 < K; k0 += 64) {
        // loads for chunk k0+32 in flight while chunk k0 computes
        if (k0 + 32 < K) { loadA(af1, k0 + 32); loadB(b1, k0 + 32); }
        __builtin_prefetch(a_base + k0 + 256, 0, 1);
        #pragma unroll
        for (int t = 0; t < 4; ++t)
            acc[t] = __builtin_amdgcn_wmma_f32_16x16x32_f16(
                false, af0.v, false, b0[t].v, (short)0, acc[t], false, false);
        if (k0 + 64 < K) { loadA(af0, k0 + 64); loadB(b0, k0 + 64); }
        if (k0 + 32 < K) {
            #pragma unroll
            for (int t = 0; t < 4; ++t)
                acc[t] = __builtin_amdgcn_wmma_f32_16x16x32_f16(
                    false, af1.v, false, b1[t].v, (short)0, acc[t], false, false);
        }
    }

    if (dmode == 2) {
        float* Df = (float*)D + dBatch * (size_t)b;
        #pragma unroll
        for (int t = 0; t < 4; ++t) {
            int ng = n0 + t * 16 + row;
            #pragma unroll
            for (int v = 0; v < 8; ++v) {
                int mg = m0 + hi * 8 + v;
                Df[(size_t)mg * ldd + ng] = acc[t][v] + (bias ? bias[mg] : 0.f);
            }
        }
    } else if (dmode == 1) {
        _Float16* Dh = (_Float16*)D + dBatch * (size_t)b;
        #pragma unroll
        for (int t = 0; t < 4; ++t) {
            int ng = n0 + t * 16 + row;
            v8h p;
            #pragma unroll
            for (int v = 0; v < 8; ++v) {
                int mg = m0 + hi * 8 + v;
                p[v] = (_Float16)(acc[t][v] + (bias ? bias[mg] : 0.f));
            }
            *(v8h*)(Dh + (size_t)ng * ldd + m0 + hi * 8) = p;
        }
    } else {
        _Float16* Dh = (_Float16*)D + dBatch * (size_t)b;
        #pragma unroll
        for (int t = 0; t < 4; ++t) {
            int ng = n0 + t * 16 + row;
            #pragma unroll
            for (int v = 0; v < 8; ++v) {
                int mg = m0 + hi * 8 + v;
                Dh[(size_t)mg * ldd + ng] = (_Float16)(acc[t][v] + (bias ? bias[mg] : 0.f));
            }
        }
    }
}

// ---------------------------------------------------------------------------
// Stage-2 GEMM: s[i,j] = sum_k G[k*256+i] * F[k*256+j]   (K = 4096)
// Operands are K-strided in memory; stage 64(K)x128(M) and 64(K)x32(N)
// tiles into LDS (rows are 256B contiguous), using the Tensor Data Mover
// when available (wave 0 issues the DMA, waits on TENSORcnt, then the block
// barriers). Fragments are then built with ds_load_u16 gathers.
// Block = 256 threads -> 128(M) x 32(N) of s; one wave = 16(M) x 32(N).
// ---------------------------------------------------------------------------
__global__ void gemm_chan_attn_kernel(const _Float16* __restrict__ G,
                                      const _Float16* __restrict__ F,
                                      float* __restrict__ S) {
    __shared__ _Float16 ldsG[64][128];   // 16 KiB
    __shared__ _Float16 ldsF[64][32];    //  4 KiB
    const int tid  = threadIdx.x;
    const int lane = tid & 31;
    const int wid  = tid >> 5;
    const int row  = lane & 15;
    const int hi   = lane >> 4;
    const int b    = blockIdx.z;
    const int i0b  = blockIdx.y * 128;
    const int j0   = blockIdx.x * 32;

    const _Float16* Gb = G + (size_t)b * CC * NN;
    const _Float16* Fb = F + (size_t)b * CC * NN;
    const int ml = wid * 16 + row;   // this wave's m within the 128-row block

    v8f acc[2] = {};
    for (int k0 = 0; k0 < NN; k0 += 64) {
        __syncthreads();   // previous tile fully consumed
#if USE_TDM
        if (wid == 0) {
            tdm_load_2d(Gb + (size_t)k0 * CC + i0b, &ldsG[0][0], 128, 64, CC, NN, CC);
            tdm_load_2d(Fb + (size_t)k0 * CC + j0,  &ldsF[0][0], 32, 64, CC, NN, CC);
            __builtin_amdgcn_s_wait_tensorcnt(0);
        }
#else
        for (int idx = tid; idx < 64 * 16; idx += 256) {
            int r = idx >> 4, c = (idx & 15) * 8;
            *(v8h*)&ldsG[r][c] = *(const v8h*)(Gb + (size_t)(k0 + r) * CC + i0b + c);
        }
        for (int idx = tid; idx < 64 * 4; idx += 256) {
            int r = idx >> 2, c = (idx & 3) * 8;
            *(v8h*)&ldsF[r][c] = *(const v8h*)(Fb + (size_t)(k0 + r) * CC + j0 + c);
        }
#endif
        __syncthreads();   // tile visible to all waves

        #pragma unroll
        for (int kc = 0; kc < 64; kc += 32) {
            Frag16 af;
            #pragma unroll
            for (int e = 0; e < 16; ++e) {
                int kk = kc + (e & 7) + ((e >> 3) << 4) + hi * 8;
                af.v[e] = ldsG[kk][ml];
            }
            #pragma unroll
            for (int t = 0; t < 2; ++t) {
                Frag16 bf_;
                #pragma unroll
                for (int e = 0; e < 16; ++e)
                    bf_.v[e] = ldsF[kc + (e & 7) + ((e >> 3) << 4) + hi * 8][t * 16 + row];
                acc[t] = __builtin_amdgcn_wmma_f32_16x16x32_f16(
                    false, af.v, false, bf_.v, (short)0, acc[t], false, false);
            }
        }
    }

    float* Sb = S + (size_t)b * CC * CC;
    #pragma unroll
    for (int t = 0; t < 2; ++t) {
        int jg = j0 + t * 16 + row;
        #pragma unroll
        for (int v = 0; v < 8; ++v) {
            int igv = i0b + wid * 16 + hi * 8 + v;
            Sb[(size_t)igv * CC + jg] = acc[t][v];
        }
    }
}

// ---------------------------------------------------------------------------
// Row softmax over 256 elements; one wave32 per row, 8 elems/lane.
// ---------------------------------------------------------------------------
__global__ void softmax_kernel(const float* __restrict__ S,
                               _Float16* __restrict__ Beta) {
    const int lane = threadIdx.x & 31;
    const int wid  = threadIdx.x >> 5;
    const int rowg = blockIdx.x * 8 + wid;       // rows = BB*CC = 2048
    const float* sr = S + (size_t)rowg * CC;

    float v[8];
    float mx = -3.0e38f;
    #pragma unroll
    for (int e = 0; e < 8; ++e) { v[e] = sr[lane + e * 32]; mx = fmaxf(mx, v[e]); }
    #pragma unroll
    for (int off = 16; off > 0; off >>= 1) mx = fmaxf(mx, __shfl_xor(mx, off, 32));
    float sum = 0.f;
    #pragma unroll
    for (int e = 0; e < 8; ++e) { v[e] = __expf(v[e] - mx); sum += v[e]; }
    #pragma unroll
    for (int off = 16; off > 0; off >>= 1) sum += __shfl_xor(sum, off, 32);
    float inv = 1.f / sum;

    _Float16* br = Beta + (size_t)rowg * CC;
    #pragma unroll
    for (int e = 0; e < 8; ++e) br[lane + e * 32] = (_Float16)(v[e] * inv);
}

// ---------------------------------------------------------------------------
// Host-side orchestration
// ---------------------------------------------------------------------------
extern "C" void kernel_launch(void* const* d_in, const int* in_sizes, int n_in,
                              void* d_out, int out_size, void* d_ws, size_t ws_size,
                              hipStream_t stream) {
    (void)in_sizes; (void)n_in; (void)out_size; (void)ws_size;

    const float* x  = (const float*)d_in[0];
    const float* Wf = (const float*)d_in[1];
    const float* bf = (const float*)d_in[2];
    const float* Wg = (const float*)d_in[3];
    const float* bg = (const float*)d_in[4];
    const float* Wh = (const float*)d_in[5];
    const float* bh = (const float*)d_in[6];
    const float* Wo = (const float*)d_in[7];
    const float* bo = (const float*)d_in[8];
    float* out = (float*)d_out;

    const size_t SZ_BNC = (size_t)BB * NN * CC;   // 8,388,608 elements
    const size_t SZ_W   = (size_t)CC * CC;        // 65,536 elements

    char* w = (char*)d_ws;
    _Float16* Xt16   = (_Float16*)(w);                          // 16 MiB (reused as Out16t)
    _Float16* Yf16   = (_Float16*)(w + SZ_BNC * 2);             // 16 MiB
    _Float16* Yg16   = (_Float16*)(w + SZ_BNC * 4);             // 16 MiB
    _Float16* Yh16   = (_Float16*)(w + SZ_BNC * 6);             // 16 MiB
    _Float16* Wf16   = (_Float16*)(w + SZ_BNC * 8);
    _Float16* Wg16   = Wf16 + SZ_W;
    _Float16* Wh16   = Wg16 + SZ_W;
    _Float16* Wo16   = Wh16 + SZ_W;
    float*    Sbuf   = (float*)(w + SZ_BNC * 8 + SZ_W * 8);     // 2 MiB
    _Float16* Beta16 = (_Float16*)((char*)Sbuf + (size_t)BB * CC * CC * 4);
    _Float16* Out16t = Xt16;   // alias: Xt dead after stage 1

    // Weights -> f16
    f32_to_f16_kernel<<<dim3((SZ_W + 255) / 256), 256, 0, stream>>>(Wf, Wf16, (int)SZ_W);
    f32_to_f16_kernel<<<dim3((SZ_W + 255) / 256), 256, 0, stream>>>(Wg, Wg16, (int)SZ_W);
    f32_to_f16_kernel<<<dim3((SZ_W + 255) / 256), 256, 0, stream>>>(Wh, Wh16, (int)SZ_W);
    f32_to_f16_kernel<<<dim3((SZ_W + 255) / 256), 256, 0, stream>>>(Wo, Wo16, (int)SZ_W);

    // x -> Xt (N,C) f16
    transpose_convert_kernel<<<dim3(NN / 32, CC / 32, BB), dim3(32, 8), 0, stream>>>(x, Xt16);

    // Stage 1: Y{f,g,h} = W @ X  (per batch), f16 out, layout [c*N + n]
    dim3 ggrid(NN / 64, CC / 128, BB);
    gemm_kcontig_kernel<<<ggrid, 256, 0, stream>>>(Wf16, 0, CC, Xt16, SZ_BNC / BB, CC,
                                                   Yf16, (size_t)CC * NN, NN, 0, bf, CC, NN, CC);
    gemm_kcontig_kernel<<<ggrid, 256, 0, stream>>>(Wg16, 0, CC, Xt16, SZ_BNC / BB, CC,
                                                   Yg16, (size_t)CC * NN, NN, 0, bg, CC, NN, CC);
    gemm_kcontig_kernel<<<ggrid, 256, 0, stream>>>(Wh16, 0, CC, Xt16, SZ_BNC / BB, CC,
                                                   Yh16, (size_t)CC * NN, NN, 0, bh, CC, NN, CC);

    // Stage 2: s = Gv^T @ Fv  (K = 4096), f32 out  (TDM-staged LDS tiles)
    gemm_chan_attn_kernel<<<dim3(CC / 32, CC / 128, BB), 256, 0, stream>>>(Yg16, Yf16, Sbuf);

    // Stage 3: softmax rows -> beta f16
    softmax_kernel<<<dim3(BB * CC / 8), 256, 0, stream>>>(Sbuf, Beta16);

    // Stage 4: out = beta @ Hv^T, stored transposed (N,C) f16
    gemm_kcontig_kernel<<<ggrid, 256, 0, stream>>>(Beta16, SZ_W, CC, Yh16, (size_t)CC * NN, CC,
                                                   Out16t, (size_t)NN * CC, CC, 1, nullptr, CC, NN, CC);

    // Stage 5: y = Wo @ out + bo, f32 to d_out
    gemm_kcontig_kernel<<<ggrid, 256, 0, stream>>>(Wo16, 0, CC, Out16t, (size_t)NN * CC, CC,
                                                   out, (size_t)CC * NN, NN, 2, bo, CC, NN, CC);
}
